// SimpleSDPAAttention_20598663152309
// MI455X (gfx1250) — matrix operand order repaired and checked by reading, not compiled, not run on hardware
//
#include <hip/hip_runtime.h>

// ---------------------------------------------------------------------------
// GQA causal flash attention for MI455X (gfx1250, wave32).
//  * QK^T and P*V on v_wmma_f32_16x16x32_f16 (f32 accum).
//  * K/V tiles DMA'd into LDS by the Tensor Data Mover (tensor_load_to_lds),
//    double-buffered K + in-iteration V, synchronized with s_wait_tensorcnt.
//  * TDM pad_interval/pad_amount give a 132-dword row stride -> conflict-free
//    ds_load_b128 fragment reads.
// One wave = one (q-head, 16-query-row) tile; loops over 32-key blocks.
// ---------------------------------------------------------------------------

typedef __attribute__((ext_vector_type(16))) _Float16 v16h;
typedef __attribute__((ext_vector_type(8)))  float    v8f;
typedef __attribute__((ext_vector_type(4)))  unsigned u32x4;
typedef __attribute__((ext_vector_type(8)))  int      i32x8;
typedef __attribute__((ext_vector_type(4)))  int      i32x4;

#define SEQ_LEN    2048
#define NUM_HEADS  32
#define NUM_KV     8
#define HEAD_DIM   128
#define GROUP      (NUM_HEADS / NUM_KV)     // 4 q heads per kv head
#define Q_STRIDE   (NUM_HEADS * HEAD_DIM)   // 4096 floats
#define KV_STRIDE  (NUM_KV * HEAD_DIM)      // 1024 floats
#define ATTN_SCALE 0.08838834764831845f
#define NEG_BIG    (-1e30f)
#define KROW       132                      // 128 + 4 dwords TDM pad (528B rows)
#define KTILE      (32 * KROW)              // floats per K/V tile buffer

union F16Frag {
    v16h     v;
    _Float16 h[16];
    unsigned u[8];
};

__device__ __forceinline__ float redmax16(float v) {
    v = fmaxf(v, __shfl_xor(v, 1, 32));
    v = fmaxf(v, __shfl_xor(v, 2, 32));
    v = fmaxf(v, __shfl_xor(v, 4, 32));
    v = fmaxf(v, __shfl_xor(v, 8, 32));
    return v;
}
__device__ __forceinline__ float redsum16(float v) {
    v += __shfl_xor(v, 1, 32);
    v += __shfl_xor(v, 2, 32);
    v += __shfl_xor(v, 4, 32);
    v += __shfl_xor(v, 8, 32);
    return v;
}

// Issue a TDM load of a 32-row x 128-float 2D tile (row stride KV_STRIDE)
// into LDS at lds_off, padded to KROW dwords per row. Rows beyond
// rows_remaining are zero-filled by TDM OOB handling.
__device__ __forceinline__ void tdm_load_tile(unsigned lds_off,
                                              const float* gptr,
                                              int rows_remaining) {
    union { u32x4 v; unsigned u[4]; } g0;
    union { i32x8 v; unsigned u[8]; } g1;
    unsigned long long ga = (unsigned long long)(size_t)gptr;

    // --- D# group 0: count=1, lds_addr, global_addr[56:0], type=2 ----------
    g0.u[0] = 1u;                                        // count = 1 (valid)
    g0.u[1] = lds_off;                                   // LDS byte address
    g0.u[2] = (unsigned)(ga & 0xFFFFFFFFull);            // global addr lo
    g0.u[3] = (unsigned)((ga >> 32) & 0x01FFFFFFull)     // global addr hi
            | (2u << 30);                                // type = 2 (image)

    // --- D# group 1: data_size=4B, pad 16B every 512B, dims/strides --------
    unsigned td0 = HEAD_DIM;                 // tensor dim0 (d)
    unsigned td1 = (unsigned)rows_remaining; // tensor dim1 (t) -> OOB zeros
    g1.u[0] = (2u << 16)                     // data_size = 4 bytes
            | (1u << 20)                     // pad_enable
            | (6u << 22)                     // pad_interval: 128 dwords (512B)
            | (3u << 25);                    // pad_amount: 4 dwords (16B)
    g1.u[1] = (td0 & 0xFFFFu) << 16;                       // tensor_dim0 lo
    g1.u[2] = ((td0 >> 16) & 0xFFFFu) | ((td1 & 0xFFFFu) << 16);
    g1.u[3] = ((td1 >> 16) & 0xFFFFu) | ((unsigned)HEAD_DIM << 16); // tile_dim0
    g1.u[4] = 32u;                                         // tile_dim1=32, dim2=0
    g1.u[5] = (unsigned)KV_STRIDE;                         // dim0 stride lo32
    g1.u[6] = 0u;                                          // stride hi / dim1 stride
    g1.u[7] = 0u;

    i32x4 z4 = {0, 0, 0, 0};                 // groups 2/3 unused (2D tile)
    i32x8 z8 = {0, 0, 0, 0, 0, 0, 0, 0};
    __builtin_amdgcn_tensor_load_to_lds(g0.v, g1.v, z4, z4, z8, 0);
}

__global__ __launch_bounds__(32, 1)
void sdpa_gqa_flash_wmma_tdm(const float* __restrict__ qg,
                             const float* __restrict__ kg,
                             const float* __restrict__ vg,
                             float* __restrict__ outg) {
    __shared__ __align__(16) float    sK[2][KTILE];   // K tiles (TDM, double buf)
    __shared__ __align__(16) float    sV[KTILE];      // V tile  (TDM)
    __shared__ _Float16 sQ[16 * HEAD_DIM];            // Q tile, f16 row-major
    __shared__ _Float16 sP[16 * 32];                  // P tile (C->A relayout)

    const int lane = threadIdx.x & 31;
    const int n    = lane & 15;   // fragment column (B/C/D) or row (A)
    const int hh   = lane >> 4;   // lane-half select

    const int head = blockIdx.y;
    const int kvh  = head / GROUP;
    const int q0   = blockIdx.x * 16;

    const float* kbaseg = kg + kvh * HEAD_DIM;
    const float* vbaseg = vg + kvh * HEAD_DIM;

    const int nkb = (q0 + 15) / 32 + 1;  // causal: key blocks up to diagonal

    // ---- Prologue: start TDM pipeline (K0, V0, K1) -------------------------
    tdm_load_tile((unsigned)(size_t)&sK[0][0], kbaseg, SEQ_LEN);
    tdm_load_tile((unsigned)(size_t)&sV[0],    vbaseg, SEQ_LEN);
    tdm_load_tile((unsigned)(size_t)&sK[1][0], kbaseg + (size_t)32 * KV_STRIDE,
                  SEQ_LEN - 32);

    // ---- Stage Q tile (16 x 128) fp32 -> f16 row-major in LDS --------------
    for (int i = lane; i < 16 * (HEAD_DIM / 4); i += 32) {
        const int row = i >> 5;
        const int u4  = i & 31;
        float4 f = *(const float4*)(qg + (size_t)(q0 + row) * Q_STRIDE
                                       + head * HEAD_DIM + u4 * 4);
        _Float16* d = sQ + row * HEAD_DIM + u4 * 4;
        d[0] = (_Float16)f.x; d[1] = (_Float16)f.y;
        d[2] = (_Float16)f.z; d[3] = (_Float16)f.w;
    }

    // ---- Q A-fragments -> registers (contiguous k = 16*hh + s, M=n) -------------
    F16Frag aQ[4];
    {
        const unsigned* qrow = (const unsigned*)(sQ + n * HEAD_DIM);
#pragma unroll
        for (int c = 0; c < 4; ++c)
#pragma unroll
            for (int p = 0; p < 8; ++p) {
                const int kk = c * 32 + hh * 16 + p * 2;
                aQ[c].u[p] = qrow[kk >> 1];
            }
    }

    // ---- Flash-attention state --------------------------------------------
    v8f accO[8];
#pragma unroll
    for (int dc = 0; dc < 8; ++dc)
#pragma unroll
        for (int i = 0; i < 8; ++i) accO[dc][i] = 0.0f;

    float rm[8], rl[8];
#pragma unroll
    for (int i = 0; i < 8; ++i) { rm[i] = NEG_BIG; rl[i] = 0.0f; }

    for (int kb = 0; kb < nkb; ++kb) {
        const int t0 = kb * 32;

        // K(kb) ready when at most {V(kb), K(kb+1)} remain outstanding.
        __builtin_amdgcn_s_wait_tensorcnt(2);
        asm volatile("" ::: "memory");

        // ---- S = Q * K^T : 16x32 scores via 8 WMMAs ------------------------
        const float* kt = sK[kb & 1];
        v8f accS[2];
#pragma unroll
        for (int s = 0; s < 2; ++s)
#pragma unroll
            for (int i = 0; i < 8; ++i) accS[s][i] = 0.0f;

#pragma unroll
        for (int c = 0; c < 4; ++c) {
#pragma unroll
            for (int s = 0; s < 2; ++s) {
                // B fragment (K^T): column = key (row of tile), 16 contiguous d
                const float* kr = kt + (s * 16 + n) * KROW + c * 32 + hh * 16;
                float4 f0 = *(const float4*)(kr + 0);
                float4 f1 = *(const float4*)(kr + 4);
                float4 f2 = *(const float4*)(kr + 8);
                float4 f3 = *(const float4*)(kr + 12);
                F16Frag b;
                b.h[0]  = (_Float16)f0.x; b.h[1]  = (_Float16)f0.y;
                b.h[2]  = (_Float16)f0.z; b.h[3]  = (_Float16)f0.w;
                b.h[4]  = (_Float16)f1.x; b.h[5]  = (_Float16)f1.y;
                b.h[6]  = (_Float16)f1.z; b.h[7]  = (_Float16)f1.w;
                b.h[8]  = (_Float16)f2.x; b.h[9]  = (_Float16)f2.y;
                b.h[10] = (_Float16)f2.z; b.h[11] = (_Float16)f2.w;
                b.h[12] = (_Float16)f3.x; b.h[13] = (_Float16)f3.y;
                b.h[14] = (_Float16)f3.z; b.h[15] = (_Float16)f3.w;
                accS[s] = __builtin_amdgcn_wmma_f32_16x16x32_f16(
                    false, aQ[c].v, false, b.v, (short)0, accS[s], false, false);
            }
        }

        // ---- Online softmax (rows i+8*hh across the 16-lane group) ---------
#pragma unroll
        for (int i = 0; i < 8; ++i) {
            const int row = q0 + i + 8 * hh;
            float s0 = accS[0][i] * ATTN_SCALE;
            float s1 = accS[1][i] * ATTN_SCALE;
            if (t0 + n      > row) s0 = NEG_BIG;
            if (t0 + 16 + n > row) s1 = NEG_BIG;

            float mx   = redmax16(fmaxf(s0, s1));
            float mnew = fmaxf(rm[i], mx);
            float corr = __expf(rm[i] - mnew);
            float p0   = __expf(s0 - mnew);
            float p1   = __expf(s1 - mnew);
            float sum  = redsum16(p0 + p1);
            rl[i] = rl[i] * corr + sum;
            rm[i] = mnew;
#pragma unroll
            for (int dc = 0; dc < 8; ++dc) accO[dc][i] *= corr;

            sP[(i + 8 * hh) * 32 + n]      = (_Float16)p0;
            sP[(i + 8 * hh) * 32 + 16 + n] = (_Float16)p1;
        }

        // ---- Reload P as A fragment (16x32, k = key index) -----------------
        F16Frag aP;
        {
            const unsigned* prow = (const unsigned*)(sP + n * 32);
#pragma unroll
            for (int p = 0; p < 8; ++p) {
                const int kk = hh * 16 + p * 2;
                aP.u[p] = prow[kk >> 1];
            }
        }

        // V(kb) ready when at most {K(kb+1)} remains outstanding.
        __builtin_amdgcn_s_wait_tensorcnt(1);
        asm volatile("" ::: "memory");

        // ---- O += P * V : 8 WMMAs over 8 d-chunks --------------------------
#pragma unroll
        for (int dc = 0; dc < 8; ++dc) {
            F16Frag bV;
#pragma unroll
            for (int p = 0; p < 8; ++p) {
                float fa = sV[(hh * 16 + 2 * p + 0) * KROW + dc * 16 + n];
                float fb = sV[(hh * 16 + 2 * p + 1) * KROW + dc * 16 + n];
                bV.h[2 * p + 0] = (_Float16)fa;
                bV.h[2 * p + 1] = (_Float16)fb;
            }
            accO[dc] = __builtin_amdgcn_wmma_f32_16x16x32_f16(
                false, aP.v, false, bV.v, (short)0, accO[dc], false, false);
        }

        // ---- Keep the TDM pipeline full: V(kb+1), K(kb+2) ------------------
        // (Clamped at the tail so wait counts stay uniform; extra tiles are
        //  written into already-consumed buffers and never read.)
        asm volatile("" ::: "memory");
        {
            int tv = t0 + 32; if (tv > SEQ_LEN - 32) tv = SEQ_LEN - 32;
            int tk = t0 + 64; if (tk > SEQ_LEN - 32) tk = SEQ_LEN - 32;
            tdm_load_tile((unsigned)(size_t)&sV[0],
                          vbaseg + (size_t)tv * KV_STRIDE, SEQ_LEN - tv);
            tdm_load_tile((unsigned)(size_t)&sK[kb & 1][0],
                          kbaseg + (size_t)tk * KV_STRIDE, SEQ_LEN - tk);
        }
    }

    // ---- Normalize and write out (fp32) ------------------------------------
#pragma unroll
    for (int i = 0; i < 8; ++i) {
        const float inv = 1.0f / rl[i];
        const int row = q0 + i + 8 * hh;
#pragma unroll
        for (int dc = 0; dc < 8; ++dc) {
            outg[(size_t)row * Q_STRIDE + head * HEAD_DIM + dc * 16 + n] =
                accO[dc][i] * inv;
        }
    }
}

extern "C" void kernel_launch(void* const* d_in, const int* in_sizes, int n_in,
                              void* d_out, int out_size, void* d_ws, size_t ws_size,
                              hipStream_t stream) {
    const float* q = (const float*)d_in[0];
    const float* k = (const float*)d_in[1];
    const float* v = (const float*)d_in[2];
    float* out = (float*)d_out;

    dim3 grid(SEQ_LEN / 16, NUM_HEADS);   // 128 x 32 blocks
    dim3 block(32);                       // one wave per block
    sdpa_gqa_flash_wmma_tdm<<<grid, block, 0, stream>>>(q, k, v, out);
}
